// CompressedAttention_88433376624960
// MI455X (gfx1250) — matrix operand order repaired and checked by reading, not compiled, not run on hardware
//
#include <hip/hip_runtime.h>

// ---------------------------------------------------------------------------
// MI455X (gfx1250, wave32) implementation.
//
// Roofline: ~343 GFLOP of GEMM vs ~0.6 GB of traffic (~25us floor at
// 23.3 TB/s) => matmul-compute-bound. All matmuls use
// v_wmma_f32_16x16x32_bf16 (bf16 operands, f32 accumulate). Tile staging
// uses CDNA5 GLOBAL_LOAD_ASYNC_TO_LDS_B128 (ASYNCcnt-tracked, no VGPR
// round-trip) with double-buffered LDS so the next K-step's tiles stream in
// while the current one is multiplied.
// ---------------------------------------------------------------------------

typedef unsigned short u16;
typedef unsigned int   u32;
typedef unsigned long long u64;
typedef __attribute__((ext_vector_type(16))) __bf16 v16bf;
typedef __attribute__((ext_vector_type(8)))  float  v8f;

#define NB    4
#define TM    4096
#define CDIM  2048
#define NH    16
#define NKV   8
#define TW    1024
#define DD    128
#define TCMP  2048
#define NSEL  512
#define OLEN  2560
#define MROWS 8192              // NB*TCMP
#define KGATE 4096              // 2*CDIM
#define NHID  2048
#define SCALE 0.088388347648318447f   // 128^-0.5
#define PAIR_FLAG (1 << 30)

// ----------------------------- helpers -------------------------------------

static __device__ __forceinline__ u16 f2bf(float x) {
  // round-to-nearest-even f32 -> bf16
  u32 u = __float_as_uint(x);
  u32 r = u + 0x7fffu + ((u >> 16) & 1u);
  return (u16)(r >> 16);
}

static __device__ __forceinline__ float silu(float x) {
  // x * sigmoid(x); v_rcp_f32 instead of the IEEE-division macro sequence
  return x * __builtin_amdgcn_rcpf(1.0f + __expf(-x));
}

static __device__ __forceinline__ v8f zero8() {
  v8f z = {0.f, 0.f, 0.f, 0.f, 0.f, 0.f, 0.f, 0.f};
  return z;
}

static __device__ __forceinline__ v8f wmma_bf16(v16bf a, v16bf b, v8f c) {
  // (neg_a, A, neg_b, B, c_mod, C, reuse_a, reuse_b)
  return __builtin_amdgcn_wmma_f32_16x16x32_bf16(false, a, false, b,
                                                 (short)0, c, false, false);
}

// CDNA5 async global->LDS copy, 16 bytes per lane. GV mode:
//   LDS offset from one VGPR, global address from a 64-bit VGPR pair.
// Tracked with ASYNCcnt; completion in issue order.
static __device__ __forceinline__ void async_copy16(const u16* lds,
                                                    const u16* gptr) {
  u32 l = (u32)(u64)lds;          // low 32 bits of a flat LDS pointer == LDS offset
  u64 g = (u64)gptr;
  asm volatile("global_load_async_to_lds_b128 %0, %1, off"
               :: "v"(l), "v"(g) : "memory");
}

// Wait until this wave's ASYNCcnt <= n (n immediate).
#define WAIT_ASYNC(n) asm volatile("s_wait_asynccnt " #n ::: "memory")

// A fragment (16x32 bf16). ISA layout (CDNA5 ISA, 16-bit A 16x32):
// lane L: row = row0 + L%16, half = L/16; VGPR v holds K pair
// {2v + 8*half (+8 if v>=4)}. Pairs are contiguous u16 -> one u32 load each.
static __device__ __forceinline__ v16bf frag_a(const u16* p, int ld,
                                               int row0, int k0) {
  union { v16bf v; u32 u[8]; } f;
  int lane = threadIdx.x & 31;
  int row  = row0 + (lane & 15);
  int half = lane >> 4;
#pragma unroll
  for (int v = 0; v < 8; ++v) {
    int k = k0 + 2 * v + 8 * half + ((v >= 4) ? 8 : 0);
    f.u[v] = *(const u32*)(p + (size_t)row * ld + k);
  }
  return f.v;
}

// B fragment (32x16 bf16). lane L: col n = n0 + L%16, half = L/16;
// VGPR v holds contraction pair {2v + 8*half (+8 if v>=4)}. Source buffer is [n][k]
// row-major (W matrices are [out,in]; K tiles are [k,d]).
static __device__ __forceinline__ v16bf frag_b(const u16* p, int ld,
                                               int n0, int k0) {
  union { v16bf v; u32 u[8]; } f;
  int lane = threadIdx.x & 31;
  int n    = n0 + (lane & 15);
  int half = lane >> 4;
#pragma unroll
  for (int v = 0; v < 8; ++v) {
    int k = k0 + 2 * v + 8 * half + ((v >= 4) ? 8 : 0);
    f.u[v] = *(const u32*)(p + (size_t)n * ld + k);
  }
  return f.v;
}

// ----------------------------- kernels -------------------------------------

__global__ __launch_bounds__(256) void cvt_kernel(const float* __restrict__ s,
                                                  u16* __restrict__ d, long n) {
  long i = (long)blockIdx.x * blockDim.x + threadIdx.x;
  long stride = (long)gridDim.x * blockDim.x;
  for (; i < n; i += stride) d[i] = f2bf(s[i]);
}

// Fused gate+up GEMM: H = silu(Xp*Wg^T) * (Xp*Wu^T), bf16 out.
// Block tile 128x128, 8 waves in 4(M)x2(N), wave tile 32x64.
// Double-buffered async LDS staging: 6 async b128 loads/thread/stage.
__global__ __launch_bounds__(256) void gemm_gateup_kernel(
    const u16* __restrict__ A, const u16* __restrict__ Bg,
    const u16* __restrict__ Bu, u16* __restrict__ Hout) {
  __shared__ __align__(16) u16 As [2][128 * 40];
  __shared__ __align__(16) u16 Bgs[2][128 * 40];
  __shared__ __align__(16) u16 Bus[2][128 * 40];
  const int K = KGATE;
  int tid = threadIdx.x, lane = tid & 31, wave = tid >> 5;
  int wm = wave & 3, wn = wave >> 2;
  int m0 = blockIdx.x * 128, n0 = blockIdx.y * 128;
  int wrow = wm * 32, wcol = wn * 64;

  v8f ag[2][4], au[2][4];
#pragma unroll
  for (int i = 0; i < 2; ++i)
#pragma unroll
    for (int j = 0; j < 4; ++j) { ag[i][j] = zero8(); au[i][j] = zero8(); }

  auto issue = [&](int buf, int k0) {
#pragma unroll
    for (int i = 0; i < 2; ++i) {   // 512 16B chunks per tile / 256 threads
      int idx = tid + i * 256, r = idx >> 2, c8 = (idx & 3) * 8;
      async_copy16(&As [buf][r * 40 + c8], &A [(size_t)(m0 + r) * K + k0 + c8]);
      async_copy16(&Bgs[buf][r * 40 + c8], &Bg[(size_t)(n0 + r) * K + k0 + c8]);
      async_copy16(&Bus[buf][r * 40 + c8], &Bu[(size_t)(n0 + r) * K + k0 + c8]);
    }
  };

  issue(0, 0);
  int s = 0;
  for (int k0 = 0; k0 < K; k0 += 32, s ^= 1) {
    if (k0 + 32 < K) { issue(s ^ 1, k0 + 32); WAIT_ASYNC(6); }
    else             { WAIT_ASYNC(0); }
    __syncthreads();                 // stage s fully landed for all waves
    v16bf a0 = frag_a(As[s], 40, wrow,      0);
    v16bf a1 = frag_a(As[s], 40, wrow + 16, 0);
#pragma unroll
    for (int n = 0; n < 4; ++n) {
      v16bf bg = frag_b(Bgs[s], 40, wcol + n * 16, 0);
      v16bf bu = frag_b(Bus[s], 40, wcol + n * 16, 0);
      ag[0][n] = wmma_bf16(a0, bg, ag[0][n]);
      ag[1][n] = wmma_bf16(a1, bg, ag[1][n]);
      au[0][n] = wmma_bf16(a0, bu, au[0][n]);
      au[1][n] = wmma_bf16(a1, bu, au[1][n]);
    }
    __syncthreads();                 // everyone done reading buf s before reuse
  }
  // C layout: VGPR r -> row r (lanes 0-15) / row 8+r (lanes 16-31), col = lane%16
  int lr = lane & 15, lh = lane >> 4;
#pragma unroll
  for (int sm = 0; sm < 2; ++sm)
#pragma unroll
    for (int n = 0; n < 4; ++n)
#pragma unroll
      for (int r = 0; r < 8; ++r) {
        int row = m0 + wrow + sm * 16 + r + 8 * lh;
        int col = n0 + wcol + n * 16 + lr;
        float g = ag[sm][n][r], u = au[sm][n][r];
        Hout[(size_t)row * NHID + col] = f2bf(silu(g) * u);
      }
}

// Down GEMM: Y = silu(H*Wd^T), f32 out. 4 async b128 loads/thread/stage.
__global__ __launch_bounds__(256) void gemm_down_kernel(
    const u16* __restrict__ A, const u16* __restrict__ Bd,
    float* __restrict__ Y) {
  __shared__ __align__(16) u16 As[2][128 * 40];
  __shared__ __align__(16) u16 Bs[2][128 * 40];
  const int K = NHID;
  int tid = threadIdx.x, lane = tid & 31, wave = tid >> 5;
  int wm = wave & 3, wn = wave >> 2;
  int m0 = blockIdx.x * 128, n0 = blockIdx.y * 128;
  int wrow = wm * 32, wcol = wn * 64;

  v8f acc[2][4];
#pragma unroll
  for (int i = 0; i < 2; ++i)
#pragma unroll
    for (int j = 0; j < 4; ++j) acc[i][j] = zero8();

  auto issue = [&](int buf, int k0) {
#pragma unroll
    for (int i = 0; i < 2; ++i) {
      int idx = tid + i * 256, r = idx >> 2, c8 = (idx & 3) * 8;
      async_copy16(&As[buf][r * 40 + c8], &A [(size_t)(m0 + r) * K + k0 + c8]);
      async_copy16(&Bs[buf][r * 40 + c8], &Bd[(size_t)(n0 + r) * K + k0 + c8]);
    }
  };

  issue(0, 0);
  int s = 0;
  for (int k0 = 0; k0 < K; k0 += 32, s ^= 1) {
    if (k0 + 32 < K) { issue(s ^ 1, k0 + 32); WAIT_ASYNC(4); }
    else             { WAIT_ASYNC(0); }
    __syncthreads();
    v16bf a0 = frag_a(As[s], 40, wrow,      0);
    v16bf a1 = frag_a(As[s], 40, wrow + 16, 0);
#pragma unroll
    for (int n = 0; n < 4; ++n) {
      v16bf b = frag_b(Bs[s], 40, wcol + n * 16, 0);
      acc[0][n] = wmma_bf16(a0, b, acc[0][n]);
      acc[1][n] = wmma_bf16(a1, b, acc[1][n]);
    }
    __syncthreads();
  }
  int lr = lane & 15, lh = lane >> 4;
#pragma unroll
  for (int sm = 0; sm < 2; ++sm)
#pragma unroll
    for (int n = 0; n < 4; ++n)
#pragma unroll
      for (int r = 0; r < 8; ++r) {
        int row = m0 + wrow + sm * 16 + r + 8 * lh;
        int col = n0 + wcol + n * 16 + lr;
        Y[(size_t)row * CDIM + col] = silu(acc[sm][n][r]);
      }
}

// Importance: per block = (b, h, 128 q rows). Each wave keeps its 16x128 Q
// block resident as 4 A-fragments; K is async-staged 32 rows at a time into
// double-buffered LDS and swept 3 times with WMMA (row-max, row-sumexp,
// normalized column-sum). Column sums accumulate in LDS (ds_add_f32), one
// global-atomic flush. The 1/H mean is skipped: monotone, so the subsequent
// top-k indices are unchanged.
__global__ __launch_bounds__(256) void importance_kernel(
    const u16* __restrict__ Qb, const u16* __restrict__ Kb,
    float* __restrict__ imp) {
  __shared__ __align__(16) u16 Ks[2][32 * 136];
  __shared__ float colacc[TCMP];
  int tid = threadIdx.x, lane = tid & 31, wave = tid >> 5;
  int blk = blockIdx.x;
  int qb = blk & 7;              // 8 q-blocks of 128 per head
  int h  = (blk >> 3) & 15;
  int b  = blk >> 7;
  int kv = h >> 1;               // contiguous GQA repeat: h -> h / GROUPS
  const u16* Qw = Qb + (size_t)((b * NH + h) * TW + qb * 128 + wave * 16) * DD;
  const u16* Kp = Kb + (size_t)((b * NKV + kv) * TCMP) * DD;

  for (int i = tid; i < TCMP; i += 256) colacc[i] = 0.f;

  v16bf aq[4];                   // Q 16x128 resident in VGPRs for all sweeps
#pragma unroll
  for (int dc = 0; dc < 4; ++dc) aq[dc] = frag_a(Qw, DD, 0, dc * 32);

  auto issueK = [&](int buf, int kb) {
#pragma unroll
    for (int i = 0; i < 2; ++i) {
      int idx = tid + i * 256, r = idx >> 4, c8 = (idx & 15) * 8;
      async_copy16(&Ks[buf][r * 136 + c8], &Kp[(size_t)(kb + r) * DD + c8]);
    }
  };

  float rmax[8], rsum[8];
#pragma unroll
  for (int r = 0; r < 8; ++r) { rmax[r] = -3.4e38f; rsum[r] = 0.f; }

  // ---- sweep 1: lane-local row max over k
  issueK(0, 0);
  int s = 0;
  for (int kb = 0; kb < TCMP; kb += 32, s ^= 1) {
    if (kb + 32 < TCMP) { issueK(s ^ 1, kb + 32); WAIT_ASYNC(2); }
    else                { WAIT_ASYNC(0); }
    __syncthreads();
#pragma unroll
    for (int kt = 0; kt < 32; kt += 16) {
      v8f c = zero8();
#pragma unroll
      for (int dc = 0; dc < 4; ++dc)
        c = wmma_bf16(aq[dc], frag_b(Ks[s], 136, kt, dc * 32), c);
#pragma unroll
      for (int r = 0; r < 8; ++r) rmax[r] = fmaxf(rmax[r], c[r] * SCALE);
    }
    __syncthreads();
  }
#pragma unroll
  for (int r = 0; r < 8; ++r) {  // reduce across the 16 k-lanes of each half
    rmax[r] = fmaxf(rmax[r], __shfl_xor(rmax[r], 1));
    rmax[r] = fmaxf(rmax[r], __shfl_xor(rmax[r], 2));
    rmax[r] = fmaxf(rmax[r], __shfl_xor(rmax[r], 4));
    rmax[r] = fmaxf(rmax[r], __shfl_xor(rmax[r], 8));
  }

  // ---- sweep 2: lane-local row sum of exp(s - max)
  issueK(0, 0);
  s = 0;
  for (int kb = 0; kb < TCMP; kb += 32, s ^= 1) {
    if (kb + 32 < TCMP) { issueK(s ^ 1, kb + 32); WAIT_ASYNC(2); }
    else                { WAIT_ASYNC(0); }
    __syncthreads();
#pragma unroll
    for (int kt = 0; kt < 32; kt += 16) {
      v8f c = zero8();
#pragma unroll
      for (int dc = 0; dc < 4; ++dc)
        c = wmma_bf16(aq[dc], frag_b(Ks[s], 136, kt, dc * 32), c);
#pragma unroll
      for (int r = 0; r < 8; ++r) rsum[r] += __expf(c[r] * SCALE - rmax[r]);
    }
    __syncthreads();
  }
  float rinv[8];
#pragma unroll
  for (int r = 0; r < 8; ++r) {
    rsum[r] += __shfl_xor(rsum[r], 1);
    rsum[r] += __shfl_xor(rsum[r], 2);
    rsum[r] += __shfl_xor(rsum[r], 4);
    rsum[r] += __shfl_xor(rsum[r], 8);
    rinv[r] = __builtin_amdgcn_rcpf(rsum[r]);
  }

  // ---- sweep 3: normalized weights, column-sum over this wave's 16 q rows
  issueK(0, 0);
  s = 0;
  for (int kb = 0; kb < TCMP; kb += 32, s ^= 1) {
    if (kb + 32 < TCMP) { issueK(s ^ 1, kb + 32); WAIT_ASYNC(2); }
    else                { WAIT_ASYNC(0); }
    __syncthreads();
#pragma unroll
    for (int kt = 0; kt < 32; kt += 16) {
      v8f c = zero8();
#pragma unroll
      for (int dc = 0; dc < 4; ++dc)
        c = wmma_bf16(aq[dc], frag_b(Ks[s], 136, kt, dc * 32), c);
      float cp = 0.f;
#pragma unroll
      for (int r = 0; r < 8; ++r)
        cp += __expf(c[r] * SCALE - rmax[r]) * rinv[r];
      cp += __shfl_xor(cp, 16);  // combine the two 8-row halves
      if ((lane >> 4) == 0)
        atomicAdd(&colacc[kb + kt + (lane & 15)], cp);
    }
    __syncthreads();
  }
  __syncthreads();
  for (int i = tid; i < TCMP; i += 256)
    atomicAdd(&imp[(size_t)b * TCMP + i], colacc[i]);
}

// Top-512 per batch, value-descending with lowest-index tie-break
// (matches jax.lax.top_k / torch.topk ordering).
__global__ __launch_bounds__(256) void topk_kernel(const float* __restrict__ imp,
                                                   int* __restrict__ sel) {
  __shared__ float vals[TCMP];
  __shared__ float rv[256];
  __shared__ int   ri[256];
  int b = blockIdx.x, tid = threadIdx.x;
  for (int i = tid; i < TCMP; i += 256) vals[i] = imp[(size_t)b * TCMP + i];
  __syncthreads();
  for (int k = 0; k < NSEL; ++k) {
    float bv = -3.4e38f; int bi = TCMP;
    for (int j = 0; j < 8; ++j) {
      int t = tid * 8 + j;
      float v = vals[t];
      if (v > bv) { bv = v; bi = t; }   // strict > keeps lowest index on tie
    }
    rv[tid] = bv; ri[tid] = bi;
    __syncthreads();
    for (int s = 128; s > 0; s >>= 1) {
      if (tid < s) {
        float ov = rv[tid + s]; int oi = ri[tid + s];
        if (ov > rv[tid] || (ov == rv[tid] && oi < ri[tid])) {
          rv[tid] = ov; ri[tid] = oi;
        }
      }
      __syncthreads();
    }
    if (tid == 0) { sel[b * NSEL + k] = ri[0]; vals[ri[0]] = -3.4e38f; }
    __syncthreads();
  }
}

// Build per-output-row source map (one block per batch).
// Faithful to the reference quirk: the k-th *value-ordered* pair lands at the
// position of the k-th *sequence-ordered* expanded slot.
__global__ __launch_bounds__(256) void plan_kernel(const int* __restrict__ sel,
                                                   int* __restrict__ srcmap) {
  __shared__ int mask[TCMP];
  __shared__ int start[TCMP];
  __shared__ int rank[TCMP];
  __shared__ int posk[NSEL];
  __shared__ int scanA[256];
  __shared__ int scanB[256];
  int b = blockIdx.x, tid = threadIdx.x;
  for (int t = tid; t < TCMP; t += 256) mask[t] = 0;
  __syncthreads();
  for (int k = tid; k < NSEL; k += 256) mask[sel[b * NSEL + k]] = 1;
  __syncthreads();
  int lm = 0;
  for (int j = 0; j < 8; ++j) lm += mask[tid * 8 + j];
  scanA[tid] = 8 + lm;   // sum of sizes (1 or 2) over this thread's 8 slots
  scanB[tid] = lm;       // sum of mask
  __syncthreads();
  if (tid == 0) {
    int a = 0, c = 0;
    for (int i = 0; i < 256; ++i) {
      int ta = scanA[i], tb = scanB[i];
      scanA[i] = a; scanB[i] = c;
      a += ta; c += tb;
    }
  }
  __syncthreads();
  int sa = scanA[tid], sb = scanB[tid];
  for (int j = 0; j < 8; ++j) {
    int t = tid * 8 + j;
    start[t] = sa; rank[t] = sb;
    int m = mask[t];
    sa += 1 + m; sb += m;
  }
  __syncthreads();
  for (int t = tid; t < TCMP; t += 256) {
    if (!mask[t]) srcmap[b * OLEN + start[t]] = t;        // kept compressed row
    else          posk[rank[t]]               = start[t]; // seq-rank -> out pos
  }
  __syncthreads();
  for (int k = tid; k < NSEL; k += 256) {
    int s = sel[b * NSEL + k];
    int pos = posk[k];
    srcmap[b * OLEN + pos]     = (2 * s)     | PAIR_FLAG;
    srcmap[b * OLEN + pos + 1] = (2 * s + 1) | PAIR_FLAG;
  }
}

// One block per output row; every row of d_out is covered exactly once.
__global__ __launch_bounds__(128) void scatter_kernel(
    const float* __restrict__ x_m, const float* __restrict__ xm_cmp,
    const int* __restrict__ srcmap, float* __restrict__ y) {
  int row = blockIdx.x;
  int b = row / OLEN;
  int s = srcmap[row];
  const float* src = (s & PAIR_FLAG)
      ? x_m    + ((size_t)b * TM   + (s & 0xffff)) * CDIM
      : xm_cmp + ((size_t)b * TCMP +  s          ) * CDIM;
  float* dst = y + (size_t)row * CDIM;
  for (int i = threadIdx.x * 4; i < CDIM; i += 128 * 4)
    *(float4*)&dst[i] = *(const float4*)&src[i];
}

// ----------------------------- launcher ------------------------------------

extern "C" void kernel_launch(void* const* d_in, const int* in_sizes, int n_in,
                              void* d_out, int out_size, void* d_ws,
                              size_t ws_size, hipStream_t stream) {
  (void)in_sizes; (void)n_in; (void)out_size; (void)ws_size;
  const float* x_m = (const float*)d_in[0];
  const float* q_w = (const float*)d_in[1];
  const float* km  = (const float*)d_in[2];
  const float* Wg  = (const float*)d_in[3];
  const float* Wu  = (const float*)d_in[4];
  const float* Wd  = (const float*)d_in[5];
  float* out = (float*)d_out;
  char*  ws  = (char*)d_ws;

  size_t off = 0;
  auto carve = [&](size_t bytes) -> void* {
    void* p = ws + off;
    off = (off + bytes + 255) & ~(size_t)255;
    return p;
  };
  u16*   xp_bf = (u16*)carve((size_t)MROWS * KGATE * 2);        // 64 MiB
  u16*   wg_bf = (u16*)carve((size_t)NHID * KGATE * 2);         // 16 MiB
  u16*   wu_bf = (u16*)carve((size_t)NHID * KGATE * 2);         // 16 MiB
  u16*   wd_bf = (u16*)carve((size_t)CDIM * NHID * 2);          //  8 MiB
  u16*   q_bf  = (u16*)carve((size_t)NB * NH * TW * DD * 2);    // 16 MiB
  u16*   k_bf  = (u16*)carve((size_t)NB * NKV * TCMP * DD * 2); // 16 MiB
  u16*   h_bf  = (u16*)carve((size_t)MROWS * NHID * 2);         // 32 MiB
  float* xmc   = (float*)carve((size_t)MROWS * CDIM * 4);       // 64 MiB
  float* imp   = (float*)carve((size_t)NB * TCMP * 4);
  int*   sel   = (int*)carve((size_t)NB * NSEL * 4);
  int*   smap  = (int*)carve((size_t)NB * OLEN * 4);

  // f32 -> bf16 operand conversion (x_pairs is a contiguous view of x_m)
  cvt_kernel<<<2048, 256, 0, stream>>>(x_m, xp_bf, (long)MROWS * KGATE);
  cvt_kernel<<<1024, 256, 0, stream>>>(Wg, wg_bf, (long)NHID * KGATE);
  cvt_kernel<<<1024, 256, 0, stream>>>(Wu, wu_bf, (long)NHID * KGATE);
  cvt_kernel<<<512,  256, 0, stream>>>(Wd, wd_bf, (long)CDIM * NHID);
  cvt_kernel<<<1024, 256, 0, stream>>>(q_w, q_bf, (long)NB * NH * TW * DD);
  cvt_kernel<<<512,  256, 0, stream>>>(km, k_bf, (long)NB * NKV * TCMP * DD);

  // compression MLP: fused gate/up GEMM + SwiGLU, then down GEMM + silu
  gemm_gateup_kernel<<<dim3(MROWS / 128, NHID / 128), 256, 0, stream>>>(
      xp_bf, wg_bf, wu_bf, h_bf);
  gemm_down_kernel<<<dim3(MROWS / 128, CDIM / 128), 256, 0, stream>>>(
      h_bf, wd_bf, xmc);

  // importance scores + softmax column sums
  hipMemsetAsync(imp, 0, (size_t)NB * TCMP * 4, stream);
  importance_kernel<<<NB * NH * (TW / 128), 256, 0, stream>>>(q_bf, k_bf, imp);

  // selection + interleave
  topk_kernel<<<NB, 256, 0, stream>>>(imp, sel);
  plan_kernel<<<NB, 256, 0, stream>>>(sel, smap);
  scatter_kernel<<<NB * OLEN, 128, 0, stream>>>(x_m, xmc, smap, out);
}
